// MPNNLayer_25941602468401
// MI455X (gfx1250) — compile-verified, matrix operand order checked by reading, not run on hardware
//
#include <hip/hip_runtime.h>

typedef _Float16 v16h __attribute__((ext_vector_type(16)));
typedef _Float16 v8h  __attribute__((ext_vector_type(8)));
typedef _Float16 v4h  __attribute__((ext_vector_type(4)));
typedef float    v8f  __attribute__((ext_vector_type(8)));

#define B_  8
#define N_  2048
#define K_  32
#define C_  128
#define H_  128
#define C3_ 384

__device__ __forceinline__ v8f vzero8() {
  v8f v;
#pragma unroll
  for (int i = 0; i < 8; ++i) v[i] = 0.0f;
  return v;
}

// CDNA5 has a hardware V_TANH_F32 (TRANS pipe, co-executes with WMMA).
__device__ __forceinline__ float fast_tanh(float x) {
#if __has_builtin(__builtin_amdgcn_tanhf)
  return __builtin_amdgcn_tanhf(x);
#elif __has_builtin(__builtin_amdgcn_tanh_f32)
  return __builtin_amdgcn_tanh_f32(x);
#else
  // tanh(x) = 1 - 2/(e^{2x}+1) via hardware exp/rcp
  float e = __expf(2.0f * x);
#if __has_builtin(__builtin_amdgcn_rcpf)
  return 1.0f - 2.0f * __builtin_amdgcn_rcpf(e + 1.0f);
#else
  return 1.0f - 2.0f / (e + 1.0f);
#endif
#endif
}

__device__ __forceinline__ float gelu_f(float x) {
  float x3 = x * x * x;
  float t = fast_tanh(0.7978845608028654f * (x + 0.044715f * x3));
  return 0.5f * x * (1.0f + t);
}

__device__ __forceinline__ float redHalf(float v) {
  v += __shfl_xor(v, 1, 32);
  v += __shfl_xor(v, 2, 32);
  v += __shfl_xor(v, 4, 32);
  v += __shfl_xor(v, 8, 32);
  return v;
}

// Load a 16x32 f16 fragment (A or B^T) from a row-major LDS tile.
// Per CDNA5 ISA 7.12.2: lane (m,hlf) holds K = {8h..8h+7} in regs 0-3 and
// K = {16+8h..16+8h+7} in regs 4-7 -> two contiguous 16B LDS reads.
__device__ __forceinline__ v16h ldfrag(const _Float16* base, int rowBase,
                                       int stride, int kOff, int m, int hlf) {
  const _Float16* p = base + (rowBase + m) * stride + kOff + 8 * hlf;
  v8h lo = *(const v8h*)p;
  v8h hi = *(const v8h*)(p + 16);
  v16h r;
#pragma unroll
  for (int i = 0; i < 8; ++i) { r[i] = lo[i]; r[i + 8] = hi[i]; }
  return r;
}

__device__ __forceinline__ v8f wmma16(v16h a, v16h b, v8f c) {
  return __builtin_amdgcn_wmma_f32_16x16x32_f16(false, a, false, b,
                                                (short)0, c, false, false);
}

// ---------------- K0: weight convert + transpose (f32[fi][fo] -> f16[fo][fi])
__global__ void cvtT_kernel(const float* __restrict__ src,
                            _Float16* __restrict__ dst, int fi, int fo) {
  int idx = blockIdx.x * blockDim.x + threadIdx.x;
  int tot = fi * fo;
  if (idx < tot) {
    int i = idx / fo;
    int o = idx - i * fo;
    dst[o * fi + i] = (_Float16)src[idx];
  }
}

// ---------------- K1 / K3: fused 3-layer message MLP (+reduce/LN or per-row LN)
// EDGE==0: node message; sum over K edges, residual on h_V, LN -> out (h_V1)
// EDGE==1: edge update; residual on h_E per row, LN -> out (h_E)
template <int EDGE>
__global__ __launch_bounds__(256) void msg_kernel(
    const float* __restrict__ gatherV,   // [B,N,C] node features for concat
    const float* __restrict__ hE_in,     // [B,N,K,C]
    const int*   __restrict__ topo,      // [B,N,K]
    const _Float16* __restrict__ w1T,    // [H][3C] f16
    const _Float16* __restrict__ w2T,    // [H][H]
    const _Float16* __restrict__ w3T,    // [C][H]
    const float* __restrict__ b1, const float* __restrict__ b2,
    const float* __restrict__ b3,
    const float* __restrict__ gma, const float* __restrict__ bta,
    const float* __restrict__ resid,     // node: h_V ; edge: h_E
    float* __restrict__ out) {
  extern __shared__ __align__(16) _Float16 smem[];
  _Float16* sW1 = smem;            // 128*384 halves
  _Float16* sW2 = smem + 49152;    // 128*128
  _Float16* sW3 = smem + 65536;    // 128*128
  const int tid  = threadIdx.x;
  const int lane = tid & 31;
  const int wv   = tid >> 5;
  _Float16* sIn  = smem + 81920  + wv * (16 * C3_);  // per-wave 16x384
  _Float16* sHid = smem + 131072 + wv * (16 * H_);   // per-wave 16x128

  // cooperative weight staging (uint4 = 8 halves per thread-op)
  {
    const uint4* s1 = (const uint4*)w1T; uint4* d1 = (uint4*)sW1;
    for (int i = tid; i < (H_ * C3_ / 8); i += 256) d1[i] = s1[i];
    const uint4* s2 = (const uint4*)w2T; uint4* d2 = (uint4*)sW2;
    for (int i = tid; i < (H_ * H_ / 8); i += 256) d2[i] = s2[i];
    const uint4* s3 = (const uint4*)w3T; uint4* d3 = (uint4*)sW3;
    for (int i = tid; i < (C_ * H_ / 8); i += 256) d3[i] = s3[i];
  }
  __syncthreads();

  const int m   = lane & 15;
  const int hlf = lane >> 4;
  const long blockRow0 = (long)blockIdx.x * 512;

  // hoist per-column constants (col = n*16 + m)
  float vb1[8], vb2[8], vb3[8], vg[8], vbt[8];
#pragma unroll
  for (int n = 0; n < 8; ++n) {
    int col = n * 16 + m;
    vb1[n] = b1[col]; vb2[n] = b2[col]; vb3[n] = b3[col];
    vg[n] = gma[col]; vbt[n] = bta[col];
  }

  // prefetch this wave's first h_E tile (16 rows x 512B = 8KB, 256B/lane)
  {
    const float* pf = hE_in + (size_t)(blockRow0 + wv * 64) * C_;
    __builtin_prefetch(pf + lane * 64, 0, 3);
  }

  float nodeAcc[8];
#pragma unroll
  for (int n = 0; n < 8; ++n) nodeAcc[n] = 0.0f;

  for (int t = 0; t < 4; ++t) {
    const long row0 = blockRow0 + wv * 64 + t * 16;
    // prefetch next tile's h_E rows while we compute this one
    if (t < 3) {
      const float* pf = hE_in + (size_t)(row0 + 16) * C_;
      __builtin_prefetch(pf + lane * 64, 0, 3);
    }
    // ---- stage 16 input rows: [h_V | h_V[nbr] | h_E] f32 -> f16 LDS
    for (int r = 0; r < 16; ++r) {
      long row = row0 + r;
      int b = (int)(row >> 16);            // / (N*K)
      int n = (int)((row >> 5) & 2047);
      int nbr = topo[row];
      const float* pV  = gatherV + ((size_t)b * N_ + n)   * C_;
      const float* pNb = gatherV + ((size_t)b * N_ + nbr) * C_;
      const float* pE  = hE_in   + (size_t)row * C_;
      int c0 = lane * 4;
      {
        float4 f = *(const float4*)(pV + c0);
        v4h h = {(_Float16)f.x, (_Float16)f.y, (_Float16)f.z, (_Float16)f.w};
        *(v4h*)(sIn + r * C3_ + c0) = h;
      }
      {
        float4 f = *(const float4*)(pNb + c0);
        v4h h = {(_Float16)f.x, (_Float16)f.y, (_Float16)f.z, (_Float16)f.w};
        *(v4h*)(sIn + r * C3_ + 128 + c0) = h;
      }
      {
        float4 f = *(const float4*)(pE + c0);
        v4h h = {(_Float16)f.x, (_Float16)f.y, (_Float16)f.z, (_Float16)f.w};
        *(v4h*)(sIn + r * C3_ + 256 + c0) = h;
      }
    }

    v8f acc[8];
    // ---- layer 1: [16x384] @ [384x128]
#pragma unroll
    for (int n = 0; n < 8; ++n) acc[n] = vzero8();
    for (int kb = 0; kb < 12; ++kb) {
      v16h a = ldfrag(sIn, 0, C3_, kb * 32, m, hlf);
#pragma unroll
      for (int n = 0; n < 8; ++n) {
        v16h bb = ldfrag(sW1, n * 16, C3_, kb * 32, m, hlf);
        acc[n] = wmma16(a, bb, acc[n]);
      }
    }
#pragma unroll
    for (int n = 0; n < 8; ++n) {
#pragma unroll
      for (int e = 0; e < 8; ++e) {
        float v = gelu_f(acc[n][e] + vb1[n]);
        sHid[(e + 8 * hlf) * H_ + n * 16 + m] = (_Float16)v;
      }
    }
    // ---- layer 2: [16x128] @ [128x128]
#pragma unroll
    for (int n = 0; n < 8; ++n) acc[n] = vzero8();
    for (int kb = 0; kb < 4; ++kb) {
      v16h a = ldfrag(sHid, 0, H_, kb * 32, m, hlf);
#pragma unroll
      for (int n = 0; n < 8; ++n) {
        v16h bb = ldfrag(sW2, n * 16, H_, kb * 32, m, hlf);
        acc[n] = wmma16(a, bb, acc[n]);
      }
    }
#pragma unroll
    for (int n = 0; n < 8; ++n) {
#pragma unroll
      for (int e = 0; e < 8; ++e) {
        float v = gelu_f(acc[n][e] + vb2[n]);
        sHid[(e + 8 * hlf) * H_ + n * 16 + m] = (_Float16)v;
      }
    }
    // ---- layer 3: [16x128] @ [128x128] (no activation)
#pragma unroll
    for (int n = 0; n < 8; ++n) acc[n] = vzero8();
    for (int kb = 0; kb < 4; ++kb) {
      v16h a = ldfrag(sHid, 0, H_, kb * 32, m, hlf);
#pragma unroll
      for (int n = 0; n < 8; ++n) {
        v16h bb = ldfrag(sW3, n * 16, H_, kb * 32, m, hlf);
        acc[n] = wmma16(a, bb, acc[n]);
      }
    }

    if (EDGE == 0) {
      // sum message rows per node (32 rows = 2 tiles per node)
#pragma unroll
      for (int n = 0; n < 8; ++n) {
        float s = 0.0f;
#pragma unroll
        for (int e = 0; e < 8; ++e) s += acc[n][e];
        s += __shfl_xor(s, 16, 32);  // add other lane-half (rows 8..15)
        nodeAcc[n] += s;
      }
      if (t & 1) {
        long node = blockRow0 / 32 + wv * 2 + (t >> 1);
        int b = (int)(node >> 11), nn = (int)(node & 2047);
        const float* pR = resid + ((size_t)b * N_ + nn) * C_;
        float y[8];
        float sm = 0.0f, sq = 0.0f;
#pragma unroll
        for (int n = 0; n < 8; ++n) {
          int col = n * 16 + m;
          float dh = (nodeAcc[n] + 32.0f * vb3[n]) * (1.0f / 60.0f);
          float v = 0.7071f * pR[col] + dh;
          y[n] = v; sm += v; sq += v * v;
          nodeAcc[n] = 0.0f;
        }
        sm = redHalf(sm); sq = redHalf(sq);
        float mean = sm * (1.0f / 128.0f);
        float var  = sq * (1.0f / 128.0f) - mean * mean;
        float inv  = rsqrtf(var + 1e-6f);
        if (lane < 16) {
          float* pO = out + ((size_t)b * N_ + nn) * C_;
#pragma unroll
          for (int n = 0; n < 8; ++n) {
            int col = n * 16 + m;
            pO[col] = (y[n] - mean) * inv * vg[n] + vbt[n];
          }
        }
      }
    } else {
      // per-edge-row residual + LayerNorm
      float sm[8], sq[8];
#pragma unroll
      for (int e = 0; e < 8; ++e) { sm[e] = 0.0f; sq[e] = 0.0f; }
#pragma unroll
      for (int n = 0; n < 8; ++n) {
        int col = n * 16 + m;
#pragma unroll
        for (int e = 0; e < 8; ++e) {
          size_t row = (size_t)(row0 + e + 8 * hlf);
          float v = 0.7071f * resid[row * C_ + col] + acc[n][e] + vb3[n];
          acc[n][e] = v;
          sm[e] += v; sq[e] += v * v;
        }
      }
      float mean[8], inv[8];
#pragma unroll
      for (int e = 0; e < 8; ++e) {
        float s = redHalf(sm[e]);
        float q = redHalf(sq[e]);
        mean[e] = s * (1.0f / 128.0f);
        float var = q * (1.0f / 128.0f) - mean[e] * mean[e];
        inv[e] = rsqrtf(var + 1e-6f);
      }
#pragma unroll
      for (int n = 0; n < 8; ++n) {
        int col = n * 16 + m;
#pragma unroll
        for (int e = 0; e < 8; ++e) {
          size_t row = (size_t)(row0 + e + 8 * hlf);
          out[row * C_ + col] = (acc[n][e] - mean[e]) * inv[e] * vg[n] + vbt[n];
        }
      }
    }
  }
}

// ---------------- K2: position-wise FFN + LN, in-place on h_V region of d_out
__global__ __launch_bounds__(256) void ffn_kernel(
    float* __restrict__ hv,  // [B*N, C] in/out
    const _Float16* __restrict__ w1T, const _Float16* __restrict__ w2T,
    const float* __restrict__ b1, const float* __restrict__ b2,
    const float* __restrict__ gma, const float* __restrict__ bta) {
  extern __shared__ __align__(16) _Float16 smem[];
  _Float16* sW1 = smem;            // 128*128
  _Float16* sW2 = smem + 16384;
  const int tid  = threadIdx.x;
  const int lane = tid & 31;
  const int wv   = tid >> 5;
  _Float16* sX = smem + 32768 + wv * (16 * C_);
  {
    const uint4* s1 = (const uint4*)w1T; uint4* d1 = (uint4*)sW1;
    for (int i = tid; i < 2048; i += 256) d1[i] = s1[i];
    const uint4* s2 = (const uint4*)w2T; uint4* d2 = (uint4*)sW2;
    for (int i = tid; i < 2048; i += 256) d2[i] = s2[i];
  }
  __syncthreads();
  const int m = lane & 15, hlf = lane >> 4;
  const long row0 = (long)blockIdx.x * 128 + wv * 16;

  float vb1[8], vb2[8], vg[8], vbt[8];
#pragma unroll
  for (int n = 0; n < 8; ++n) {
    int col = n * 16 + m;
    vb1[n] = b1[col]; vb2[n] = b2[col]; vg[n] = gma[col]; vbt[n] = bta[col];
  }

  for (int r = 0; r < 16; ++r) {
    const float* p = hv + (size_t)(row0 + r) * C_;
    int c0 = lane * 4;
    float4 f = *(const float4*)(p + c0);
    v4h h = {(_Float16)f.x, (_Float16)f.y, (_Float16)f.z, (_Float16)f.w};
    *(v4h*)(sX + r * C_ + c0) = h;
  }

  v8f acc[8];
#pragma unroll
  for (int n = 0; n < 8; ++n) acc[n] = vzero8();
  for (int kb = 0; kb < 4; ++kb) {
    v16h a = ldfrag(sX, 0, C_, kb * 32, m, hlf);
#pragma unroll
    for (int n = 0; n < 8; ++n) {
      v16h bb = ldfrag(sW1, n * 16, C_, kb * 32, m, hlf);
      acc[n] = wmma16(a, bb, acc[n]);
    }
  }
#pragma unroll
  for (int n = 0; n < 8; ++n) {
#pragma unroll
    for (int e = 0; e < 8; ++e) {
      float v = gelu_f(acc[n][e] + vb1[n]);
      sX[(e + 8 * hlf) * C_ + n * 16 + m] = (_Float16)v;
    }
  }
#pragma unroll
  for (int n = 0; n < 8; ++n) acc[n] = vzero8();
  for (int kb = 0; kb < 4; ++kb) {
    v16h a = ldfrag(sX, 0, C_, kb * 32, m, hlf);
#pragma unroll
    for (int n = 0; n < 8; ++n) {
      v16h bb = ldfrag(sW2, n * 16, C_, kb * 32, m, hlf);
      acc[n] = wmma16(a, bb, acc[n]);
    }
  }
  // residual + per-row LN (row = node)
  float sm[8], sq[8];
#pragma unroll
  for (int e = 0; e < 8; ++e) { sm[e] = 0.0f; sq[e] = 0.0f; }
#pragma unroll
  for (int n = 0; n < 8; ++n) {
    int col = n * 16 + m;
#pragma unroll
    for (int e = 0; e < 8; ++e) {
      size_t row = (size_t)(row0 + e + 8 * hlf);
      float v = 0.7071f * hv[row * C_ + col] + acc[n][e] + vb2[n];
      acc[n][e] = v;
      sm[e] += v; sq[e] += v * v;
    }
  }
  float mean[8], inv[8];
#pragma unroll
  for (int e = 0; e < 8; ++e) {
    float s = redHalf(sm[e]);
    float q = redHalf(sq[e]);
    mean[e] = s * (1.0f / 128.0f);
    float var = q * (1.0f / 128.0f) - mean[e] * mean[e];
    inv[e] = rsqrtf(var + 1e-6f);
  }
#pragma unroll
  for (int n = 0; n < 8; ++n) {
    int col = n * 16 + m;
#pragma unroll
    for (int e = 0; e < 8; ++e) {
      size_t row = (size_t)(row0 + e + 8 * hlf);
      hv[row * C_ + col] = (acc[n][e] - mean[e]) * inv[e] * vg[n] + vbt[n];
    }
  }
}

extern "C" void kernel_launch(void* const* d_in, const int* in_sizes, int n_in,
                              void* d_out, int out_size, void* d_ws,
                              size_t ws_size, hipStream_t stream) {
  (void)in_sizes; (void)n_in; (void)out_size; (void)ws_size;
  const float* hV  = (const float*)d_in[0];
  const float* hE  = (const float*)d_in[1];
  const int*   top = (const int*)d_in[2];
  const float* wn1 = (const float*)d_in[3];  const float* bn1 = (const float*)d_in[4];
  const float* wn2 = (const float*)d_in[5];  const float* bn2 = (const float*)d_in[6];
  const float* wn3 = (const float*)d_in[7];  const float* bn3 = (const float*)d_in[8];
  const float* wf1 = (const float*)d_in[9];  const float* bf1 = (const float*)d_in[10];
  const float* wf2 = (const float*)d_in[11]; const float* bf2 = (const float*)d_in[12];
  const float* we1 = (const float*)d_in[13]; const float* be1 = (const float*)d_in[14];
  const float* we2 = (const float*)d_in[15]; const float* be2 = (const float*)d_in[16];
  const float* we3 = (const float*)d_in[17]; const float* be3 = (const float*)d_in[18];
  const float* g1  = (const float*)d_in[19]; const float* bt1 = (const float*)d_in[20];
  const float* g2  = (const float*)d_in[21]; const float* bt2 = (const float*)d_in[22];
  const float* g3  = (const float*)d_in[23]; const float* bt3 = (const float*)d_in[24];

  float* outV = (float*)d_out;                       // [B,N,C]
  float* outE = outV + (size_t)B_ * N_ * C_;         // [B,N,K,C]

  _Float16* ws  = (_Float16*)d_ws;
  _Float16* wn1T = ws;            // 128x384
  _Float16* wn2T = ws + 49152;    // 128x128
  _Float16* wn3T = ws + 65536;
  _Float16* wf1T = ws + 81920;
  _Float16* wf2T = ws + 98304;
  _Float16* we1T = ws + 114688;   // 128x384
  _Float16* we2T = ws + 163840;
  _Float16* we3T = ws + 180224;

  // K0: weight convert + transpose
  cvtT_kernel<<<192, 256, 0, stream>>>(wn1, wn1T, C3_, H_);
  cvtT_kernel<<<64,  256, 0, stream>>>(wn2, wn2T, H_, H_);
  cvtT_kernel<<<64,  256, 0, stream>>>(wn3, wn3T, H_, C_);
  cvtT_kernel<<<64,  256, 0, stream>>>(wf1, wf1T, C_, H_);
  cvtT_kernel<<<64,  256, 0, stream>>>(wf2, wf2T, H_, C_);
  cvtT_kernel<<<192, 256, 0, stream>>>(we1, we1T, C3_, H_);
  cvtT_kernel<<<64,  256, 0, stream>>>(we2, we2T, H_, H_);
  cvtT_kernel<<<64,  256, 0, stream>>>(we3, we3T, H_, C_);

  const size_t smemMsg = (size_t)147456 * sizeof(_Float16); // 288 KB
  const size_t smemFfn = (size_t)49152  * sizeof(_Float16); // 96 KB

  // K1: node message + reduce + residual + LN1 -> outV (= h_V after LN1)
  msg_kernel<0><<<1024, 256, smemMsg, stream>>>(
      hV, hE, top, wn1T, wn2T, wn3T, bn1, bn2, bn3, g1, bt1, hV, outV);

  // K2: FFN + LN2, in-place on outV
  ffn_kernel<<<128, 256, smemFfn, stream>>>(outV, wf1T, wf2T, bf1, bf2, g2, bt2);

  // K3: edge message + residual + LN3 -> outE
  msg_kernel<1><<<1024, 256, smemMsg, stream>>>(
      outV, hE, top, we1T, we2T, we3T, be1, be2, be3, g3, bt3, hE, outE);
}
